// Transformer_34763465294157
// MI455X (gfx1250) — matrix operand order repaired and checked by reading, not compiled, not run on hardware
//
#include <hip/hip_runtime.h>
#include <hip/hip_bf16.h>

// ---------------------------------------------------------------------------
// Types for CDNA5 WMMA (wave32): v_wmma_f32_16x16x32_bf16
// ---------------------------------------------------------------------------
typedef __attribute__((ext_vector_type(16))) __bf16 v16bf;
typedef __attribute__((ext_vector_type(8)))  float  v8f;

union FragU {
    v16bf bf;
    unsigned int u[8];
};

// bf16 convert, round-half-up (differs from RNE only on exact ties)
__device__ __forceinline__ unsigned short f2bf(float f) {
    return (unsigned short)((__float_as_uint(f) + 0x8000u) >> 16);
}

// pack two floats to packed bf16 pair {hi,lo} with a single v_perm_b32
__device__ __forceinline__ unsigned int pack2bf(float lo, float hi) {
    unsigned int ulo = __float_as_uint(lo) + 0x8000u;
    unsigned int uhi = __float_as_uint(hi) + 0x8000u;
    // result bytes: [hi.b3, hi.b2, lo.b3, lo.b2]  (a=uhi -> idx 4..7, b=ulo -> idx 0..3)
    return __builtin_amdgcn_perm(uhi, ulo, 0x07060302u);
}

__device__ __forceinline__ v8f wmma_bf16(v16bf a, v16bf b, v8f c) {
    return __builtin_amdgcn_wmma_f32_16x16x32_bf16(
        /*neg_a=*/false, a, /*neg_b=*/false, b,
        /*c_mod=*/(short)0, c, /*reuse_a=*/false, /*reuse_b=*/false);
}

// A-matrix fragment (16x32 bf16). rowPtr points at LDS row [m][k0..k0+31].
// lane<16:  K pairs {0..7, 16..23};  lane>=16: K pairs {8..15, 24..31}
__device__ __forceinline__ v16bf load_a_frag(const unsigned short* rowPtr, int lhi) {
    FragU f;
#pragma unroll
    for (int i = 0; i < 8; i++) {
        int k = (i < 4 ? 2 * i : 2 * i + 8) + (lhi ? 8 : 0);
        f.u[i] = *(const unsigned int*)(rowPtr + k);
    }
    return f.bf;
}

// B-matrix fragment (32x16 bf16). colPtr points at LDS "column" [n][k0..k0+31]
// (B stored K-contiguous per n). lane<16: K {0..7, 16..23}, lane>=16: K {8..15, 24..31}.
__device__ __forceinline__ v16bf load_b_frag(const unsigned short* colPtr, int lhi) {
    FragU f;
    int base = lhi ? 8 : 0;
#pragma unroll
    for (int i = 0; i < 8; i++)
        f.u[i] = *(const unsigned int*)(colPtr + base + (i < 4 ? 2 * i : 2 * i + 8));
    return f.bf;
}

__device__ __forceinline__ v8f vzero8() {
    v8f z = {0.f, 0.f, 0.f, 0.f, 0.f, 0.f, 0.f, 0.f};
    return z;
}

// ---------------------------------------------------------------------------
// Aligned GEMM: C[M,N] = act(A[M,K] @ B[K,N] + bias)   (f32 in/out, bf16 WMMA)
// REQUIRES: M % 64 == 0, N % 64 == 0, K % 32 == 0 (16B-aligned rows).
// 64x64x32 block tile, 128 threads (4 waves), each wave: 2x2 of 16x16x32 WMMA.
// Loads are issued in a batch before conversion so they overlap in flight.
// ---------------------------------------------------------------------------
#define BM 64
#define BN 64
#define BKK 32

__global__ __launch_bounds__(128) void gemm_kernel(
    const float* __restrict__ A, const float* __restrict__ Bw,
    const float* __restrict__ bias, float* __restrict__ C,
    int M, int N, int K, int relu)
{
    __shared__ unsigned short As[BM][BKK];   // [m][k]
    __shared__ unsigned short Bs[BN][BKK];   // [n][k]  (transposed)

    int tid  = threadIdx.x;
    int lane = tid & 31;
    int w    = tid >> 5;
    int lhi  = lane >> 4;
    int ln16 = lane & 15;
    int wm   = w >> 1, wn = w & 1;
    int bm   = blockIdx.y * BM, bn = blockIdx.x * BN;

    v8f acc[2][2];
    acc[0][0] = vzero8(); acc[0][1] = vzero8();
    acc[1][0] = vzero8(); acc[1][1] = vzero8();

    int ksteps = K / BKK;
    for (int ks = 0; ks < ksteps; ks++) {
        int k0 = ks * BKK;

        // ---- issue all global loads for this K-step first ----
        float4 va[4];                        // A: 64x32 = 512 float4, 4/thread
#pragma unroll
        for (int i = 0; i < 4; i++) {
            int li = i * 128 + tid;
            int r = li >> 3, c4 = li & 7;
            va[i] = *(const float4*)(A + (size_t)(bm + r) * K + k0 + c4 * 4);
        }
        float4 vb0[2], vb1[2];               // B: 2(k) x 4(n) micro-blocks, 2/thread
#pragma unroll
        for (int i = 0; i < 2; i++) {
            int p = i * 128 + tid;           // 0..255
            int k2 = p >> 4, c4 = p & 15;    // k-pair, n/4
            const float* bp = Bw + (size_t)(k0 + k2 * 2) * N + bn + c4 * 4;
            vb0[i] = *(const float4*)(bp);
            vb1[i] = *(const float4*)(bp + N);
        }

        // speculative prefetch of next K-tile (gfx1250 global_prefetch_b8)
        if (ks + 1 < ksteps) {
            int kn = k0 + BKK;
            __builtin_prefetch(A + (size_t)(bm + (tid >> 1)) * K + kn + (tid & 1) * 16, 0, 1);
            __builtin_prefetch(Bw + (size_t)(kn + (tid >> 2)) * N + bn + (tid & 3) * 16, 0, 1);
        }

        // ---- convert + store to LDS ----
#pragma unroll
        for (int i = 0; i < 4; i++) {
            int li = i * 128 + tid;
            int r = li >> 3, c4 = li & 7;
            uint2 p;
            p.x = pack2bf(va[i].x, va[i].y);
            p.y = pack2bf(va[i].z, va[i].w);
            *(uint2*)(&As[r][c4 * 4]) = p;
        }
#pragma unroll
        for (int i = 0; i < 2; i++) {
            int p = i * 128 + tid;
            int k2 = p >> 4, c4 = p & 15;
            int k = k2 * 2, n0 = c4 * 4;
            *(unsigned int*)(&Bs[n0 + 0][k]) = pack2bf(vb0[i].x, vb1[i].x);
            *(unsigned int*)(&Bs[n0 + 1][k]) = pack2bf(vb0[i].y, vb1[i].y);
            *(unsigned int*)(&Bs[n0 + 2][k]) = pack2bf(vb0[i].z, vb1[i].z);
            *(unsigned int*)(&Bs[n0 + 3][k]) = pack2bf(vb0[i].w, vb1[i].w);
        }
        __syncthreads();

        v16bf af0 = load_a_frag(&As[wm * 32 + ln16][0], lhi);
        v16bf af1 = load_a_frag(&As[wm * 32 + 16 + ln16][0], lhi);
        v16bf bf0 = load_b_frag(&Bs[wn * 32 + ln16][0], lhi);
        v16bf bf1 = load_b_frag(&Bs[wn * 32 + 16 + ln16][0], lhi);

        acc[0][0] = wmma_bf16(af0, bf0, acc[0][0]);
        acc[0][1] = wmma_bf16(af0, bf1, acc[0][1]);
        acc[1][0] = wmma_bf16(af1, bf0, acc[1][0]);
        acc[1][1] = wmma_bf16(af1, bf1, acc[1][1]);
        __syncthreads();
    }

#pragma unroll
    for (int tm = 0; tm < 2; tm++)
#pragma unroll
        for (int tn = 0; tn < 2; tn++) {
            int col = bn + wn * 32 + tn * 16 + ln16;
            float bc = bias ? bias[col] : 0.f;
#pragma unroll
            for (int r = 0; r < 8; r++) {
                int row = bm + wm * 32 + tm * 16 + r + lhi * 8;
                float v = acc[tm][tn][r] + bc;
                if (relu) v = fmaxf(v, 0.f);
                C[(size_t)row * N + col] = v;
            }
        }
}

// ---------------------------------------------------------------------------
// Flash attention: O = softmax(mask(QK^T/sqrt(dk))) V,  DK=64, 64-query tiles.
// grid = (S/64, B*H), 128 threads (4 waves). Wave owns 16 q rows x 64 keys.
// ---------------------------------------------------------------------------
__global__ __launch_bounds__(128) void attn_kernel(
    const float* __restrict__ Q, const float* __restrict__ Kv,
    const float* __restrict__ Vv, const float* __restrict__ keymask,
    float* __restrict__ O, int Bn, int Sn, int Hn, int causal)
{
    __shared__ unsigned short Qs[64][64];      // [q][d]
    __shared__ unsigned short Ks[64][64];      // [key][d]
    __shared__ unsigned short Vs[64][64];      // [d][key]  (transposed)
    __shared__ unsigned short Ps[4][16][64];   // per-wave P tile [q][key]

    int tid  = threadIdx.x;
    int lane = tid & 31;
    int w    = tid >> 5;
    int lhi  = lane >> 4;
    int ln16 = lane & 15;
    int qb   = blockIdx.x;
    int bh   = blockIdx.y;
    int b    = bh / Hn, hd = bh % Hn;
    int Dm   = Hn * 64;
    const float scale = 0.125f;   // 1/sqrt(64)

    // Q tile: 64x64 floats = 1024 float4, 8 per thread; batch loads then pack
    {
        float4 vq[8];
#pragma unroll
        for (int i = 0; i < 8; i++) {
            int li = i * 128 + tid;
            int r = li >> 4, c4 = li & 15;
            vq[i] = *(const float4*)(
                Q + ((size_t)(b * Sn) + qb * 64 + r) * Dm + hd * 64 + c4 * 4);
        }
#pragma unroll
        for (int i = 0; i < 8; i++) {
            int li = i * 128 + tid;
            int r = li >> 4, c4 = li & 15;
            uint2 p;
            p.x = pack2bf(vq[i].x * scale, vq[i].y * scale);
            p.y = pack2bf(vq[i].z * scale, vq[i].w * scale);
            *(uint2*)(&Qs[r][c4 * 4]) = p;
        }
    }
    __syncthreads();

    v16bf aq0 = load_a_frag(&Qs[w * 16 + ln16][0], lhi);
    v16bf aq1 = load_a_frag(&Qs[w * 16 + ln16][32], lhi);

    v8f o[4];
    o[0] = vzero8(); o[1] = vzero8(); o[2] = vzero8(); o[3] = vzero8();
    float mrow[8], lrow[8];
#pragma unroll
    for (int r = 0; r < 8; r++) { mrow[r] = -1e30f; lrow[r] = 0.f; }

    int jmax = causal ? qb : (Sn / 64 - 1);
    for (int j = 0; j <= jmax; j++) {
        // ---- issue all K/V loads for this key block first ----
        float4 vk[8];
#pragma unroll
        for (int i = 0; i < 8; i++) {
            int li = i * 128 + tid;
            int r = li >> 4, c4 = li & 15;
            vk[i] = *(const float4*)(
                Kv + ((size_t)(b * Sn) + j * 64 + r) * Dm + hd * 64 + c4 * 4);
        }
        float4 vv0[4], vv1[4];
#pragma unroll
        for (int i = 0; i < 4; i++) {
            int p = i * 128 + tid;              // 0..511
            int k2 = p >> 4, c4 = p & 15;       // key-pair, d/4
            size_t g0 = ((size_t)(b * Sn) + j * 64 + k2 * 2) * Dm + hd * 64 + c4 * 4;
            vv0[i] = *(const float4*)(Vv + g0);
            vv1[i] = *(const float4*)(Vv + g0 + Dm);
        }
        // ---- convert + store to LDS ----
#pragma unroll
        for (int i = 0; i < 8; i++) {
            int li = i * 128 + tid;
            int r = li >> 4, c4 = li & 15;
            uint2 p;
            p.x = pack2bf(vk[i].x, vk[i].y);
            p.y = pack2bf(vk[i].z, vk[i].w);
            *(uint2*)(&Ks[r][c4 * 4]) = p;
        }
#pragma unroll
        for (int i = 0; i < 4; i++) {
            int p = i * 128 + tid;
            int k2 = p >> 4, c4 = p & 15;
            int key = k2 * 2, d0 = c4 * 4;
            *(unsigned int*)(&Vs[d0 + 0][key]) = pack2bf(vv0[i].x, vv1[i].x);
            *(unsigned int*)(&Vs[d0 + 1][key]) = pack2bf(vv0[i].y, vv1[i].y);
            *(unsigned int*)(&Vs[d0 + 2][key]) = pack2bf(vv0[i].z, vv1[i].z);
            *(unsigned int*)(&Vs[d0 + 3][key]) = pack2bf(vv0[i].w, vv1[i].w);
        }
        __syncthreads();

        v8f sf[4];
        sf[0] = vzero8(); sf[1] = vzero8(); sf[2] = vzero8(); sf[3] = vzero8();
#pragma unroll
        for (int t = 0; t < 4; t++) {
            v16bf bk0 = load_b_frag(&Ks[t * 16 + ln16][0], lhi);
            sf[t] = wmma_bf16(aq0, bk0, sf[t]);
            v16bf bk1 = load_b_frag(&Ks[t * 16 + ln16][32], lhi);
            sf[t] = wmma_bf16(aq1, bk1, sf[t]);
        }

        // mask + online softmax (rows live in 16-lane halves)
#pragma unroll
        for (int r = 0; r < 8; r++) {
            int qg = qb * 64 + w * 16 + r + lhi * 8;
            float sv[4];
            float mx = -1e30f;
#pragma unroll
            for (int t = 0; t < 4; t++) {
                int kg = j * 64 + t * 16 + ln16;
                float s = sf[t][r];
                bool ok = (keymask[b * Sn + kg] > 0.5f) && (!causal || kg <= qg);
                s = ok ? s : -1e9f;
                sv[t] = s;
                mx = fmaxf(mx, s);
            }
#pragma unroll
            for (int m = 1; m < 16; m <<= 1) mx = fmaxf(mx, __shfl_xor(mx, m));
            float mnew  = fmaxf(mrow[r], mx);
            float alpha = __expf(mrow[r] - mnew);
            float rs = 0.f;
#pragma unroll
            for (int t = 0; t < 4; t++) {
                float p = __expf(sv[t] - mnew);
                rs += p;
                Ps[w][r + lhi * 8][t * 16 + ln16] = f2bf(p);
            }
#pragma unroll
            for (int m = 1; m < 16; m <<= 1) rs += __shfl_xor(rs, m);
            lrow[r] = lrow[r] * alpha + rs;
            mrow[r] = mnew;
#pragma unroll
            for (int t = 0; t < 4; t++) o[t][r] *= alpha;
        }

        // O += P @ V  (P read back from LDS in A-layout)
        v16bf pa0 = load_a_frag(&Ps[w][ln16][0], lhi);
        v16bf pa1 = load_a_frag(&Ps[w][ln16][32], lhi);
#pragma unroll
        for (int t = 0; t < 4; t++) {
            v16bf vb0 = load_b_frag(&Vs[t * 16 + ln16][0], lhi);
            o[t] = wmma_bf16(pa0, vb0, o[t]);
            v16bf vb1 = load_b_frag(&Vs[t * 16 + ln16][32], lhi);
            o[t] = wmma_bf16(pa1, vb1, o[t]);
        }
        __syncthreads();
    }

#pragma unroll
    for (int t = 0; t < 4; t++)
#pragma unroll
        for (int r = 0; r < 8; r++) {
            int row = qb * 64 + w * 16 + r + lhi * 8;
            O[((size_t)(b * Sn) + row) * Dm + hd * 64 + t * 16 + ln16] =
                o[t][r] / lrow[r];
        }
}

// ---------------------------------------------------------------------------
// LayerNorm with residual: out = LN(x + r) * g + b   (D = 512, 128 threads)
// ---------------------------------------------------------------------------
__global__ __launch_bounds__(128) void ln_kernel(
    const float* __restrict__ x, const float* __restrict__ rr,
    const float* __restrict__ g, const float* __restrict__ bb,
    float* __restrict__ out, int Dm)
{
    int row = blockIdx.x, tid = threadIdx.x;
    const float* xp = x + (size_t)row * Dm;
    const float* rp = rr + (size_t)row * Dm;
    float4 xv = *(const float4*)(xp + tid * 4);
    float4 rv = *(const float4*)(rp + tid * 4);
    float v0 = xv.x + rv.x, v1 = xv.y + rv.y, v2 = xv.z + rv.z, v3 = xv.w + rv.w;
    float s  = v0 + v1 + v2 + v3;
    float s2 = v0 * v0 + v1 * v1 + v2 * v2 + v3 * v3;

    __shared__ float rs[128], rs2[128];
    rs[tid] = s; rs2[tid] = s2;
    __syncthreads();
    for (int off = 64; off > 0; off >>= 1) {
        if (tid < off) { rs[tid] += rs[tid + off]; rs2[tid] += rs2[tid + off]; }
        __syncthreads();
    }
    float mean = rs[0] / Dm;
    float var  = rs2[0] / Dm - mean * mean;
    float rstd = rsqrtf(var + 1e-6f);

    float4 gv = *(const float4*)(g + tid * 4);
    float4 bv = *(const float4*)(bb + tid * 4);
    float4 ov;
    ov.x = (v0 - mean) * rstd * gv.x + bv.x;
    ov.y = (v1 - mean) * rstd * gv.y + bv.y;
    ov.z = (v2 - mean) * rstd * gv.z + bv.z;
    ov.w = (v3 - mean) * rstd * gv.w + bv.w;
    *(float4*)(out + (size_t)row * Dm + tid * 4) = ov;
}

// ---------------------------------------------------------------------------
// Small support kernels
// ---------------------------------------------------------------------------
__global__ void postable_kernel(float* __restrict__ tab, int Sn, int Dm) {
    int idx = blockIdx.x * blockDim.x + threadIdx.x;
    if (idx >= Sn * Dm) return;
    int p = idx / Dm, i = idx % Dm;
    float expo  = (float)(2 * (i / 2)) / (float)Dm;
    float angle = (float)p * powf(10000.0f, -expo);
    tab[idx] = (i & 1) ? cosf(angle) : sinf(angle);
}

// Wc[b,s,t] = (w0*gk[b,0,s,t] + w1*gk[b,1,s,t] + w2*gk[b,2,s,t]) / (w0+w1+w2)
__global__ void wcomb_kernel(const float* __restrict__ gk, float* __restrict__ Wc,
                             int Bn, int Sn) {
    long long idx4 = (long long)blockIdx.x * blockDim.x + threadIdx.x;
    long long tot4 = (long long)Bn * Sn * Sn / 4;
    if (idx4 >= tot4) return;
    long long ss = (long long)Sn * Sn;
    long long idx = idx4 * 4;
    int b = (int)(idx / ss);
    long long rem = idx - (long long)b * ss;
    const float w0 = 1.0f, w1 = 0.9f, w2 = 0.81f;
    float inv = 1.0f / (w0 + w1 + w2);
    size_t base = (size_t)b * 3 * ss + rem;
    float4 a = *(const float4*)(gk + base);
    float4 c = *(const float4*)(gk + base + ss);
    float4 d = *(const float4*)(gk + base + 2 * ss);
    float4 o;
    o.x = (w0 * a.x + w1 * c.x + w2 * d.x) * inv;
    o.y = (w0 * a.y + w1 * c.y + w2 * d.y) * inv;
    o.z = (w0 * a.z + w1 * c.z + w2 * d.z) * inv;
    o.w = (w0 * a.w + w1 * c.w + w2 * d.w) * inv;
    *(float4*)(Wc + idx) = o;
}

// mask[r] = 1 if no element of row equals PAD_IDX (-1.0)
__global__ void rowmask_kernel(const float* __restrict__ x, float* __restrict__ mask,
                               int rows, int width) {
    int r = blockIdx.x * blockDim.x + threadIdx.x;
    if (r >= rows) return;
    const float* p = x + (size_t)r * width;
    int ok = 1;
    for (int i = 0; i < width; i++)
        if (p[i] == -1.0f) { ok = 0; break; }
    mask[r] = ok ? 1.f : 0.f;
}

// dst[drows x dcols] = zero-padded src[srows x scols]
__global__ void pad_copy_kernel(const float* __restrict__ src, float* __restrict__ dst,
                                int srows, int scols, int drows, int dcols) {
    int idx = blockIdx.x * blockDim.x + threadIdx.x;
    if (idx >= drows * dcols) return;
    int r = idx / dcols, c = idx % dcols;
    dst[idx] = (r < srows && c < scols) ? src[(size_t)r * scols + c] : 0.f;
}

// dst[rows x dcols] = src[rows x sstride] (take first dcols of each row)
__global__ void unpad_copy_kernel(const float* __restrict__ src, float* __restrict__ dst,
                                  int rows, int sstride, int dcols) {
    int idx = blockIdx.x * blockDim.x + threadIdx.x;
    if (idx >= rows * dcols) return;
    int r = idx / dcols, c = idx % dcols;
    dst[idx] = src[(size_t)r * sstride + c];
}

// ---------------------------------------------------------------------------
// Host orchestration
// ---------------------------------------------------------------------------
static inline void launch_gemm(const float* A, const float* B, const float* bias,
                               float* C, int M, int N, int K, int relu,
                               hipStream_t s) {
    dim3 grid(N / BN, M / BM);
    gemm_kernel<<<grid, dim3(128), 0, s>>>(A, B, bias, C, M, N, K, relu);
}

static inline void launch_pad(const float* src, float* dst, int srows, int scols,
                              int drows, int dcols, hipStream_t s) {
    int n = drows * dcols;
    pad_copy_kernel<<<(n + 255) / 256, 256, 0, s>>>(src, dst, srows, scols, drows, dcols);
}

extern "C" void kernel_launch(void* const* d_in, const int* in_sizes, int n_in,
                              void* d_out, int out_size, void* d_ws, size_t ws_size,
                              hipStream_t stream) {
    (void)in_sizes; (void)n_in; (void)out_size; (void)ws_size;
    const int Bc = 2, Sc = 1024, Vc = 513, Dc = 512, Hc = 8, DFFc = 2048, Lc = 6;
    const int Vp = 576;          // 513 padded to multiple of 64
    const int Mrows = Bc * Sc;   // 2048

    const float* dec_in = (const float*)d_in[0];
    const float* prev   = (const float*)d_in[1];
    const float* gk     = (const float*)d_in[6];
    const float* emb_w1 = (const float*)d_in[7];
    const float* emb_b1 = (const float*)d_in[8];
    const float* emb_w2 = (const float*)d_in[9];
    const float* emb_b2 = (const float*)d_in[10];
    const float* emb_w3 = (const float*)d_in[11];
    const float* emb_b3 = (const float*)d_in[12];
    const float* ln0_g  = (const float*)d_in[13];
    const float* ln0_b  = (const float*)d_in[14];
    const float* wq     = (const float*)d_in[15];
    const float* wk     = (const float*)d_in[16];
    const float* wv     = (const float*)d_in[17];
    const float* wo     = (const float*)d_in[18];
    const float* cq     = (const float*)d_in[19];
    const float* ck     = (const float*)d_in[20];
    const float* cv     = (const float*)d_in[21];
    const float* co     = (const float*)d_in[22];
    const float* ffn_w1 = (const float*)d_in[23];
    const float* ffn_w2 = (const float*)d_in[24];
    const float* ffn_b1 = (const float*)d_in[25];
    const float* ffn_b2 = (const float*)d_in[26];
    const float* ln1_g  = (const float*)d_in[27];
    const float* ln1_b  = (const float*)d_in[28];
    const float* ln2_g  = (const float*)d_in[29];
    const float* ln2_b  = (const float*)d_in[30];
    const float* ln3_g  = (const float*)d_in[31];
    const float* ln3_b  = (const float*)d_in[32];
    const float* p_w1   = (const float*)d_in[33];
    const float* p_b1   = (const float*)d_in[34];
    const float* p_w2   = (const float*)d_in[35];
    const float* p_b2   = (const float*)d_in[36];
    const float* p_w3   = (const float*)d_in[37];
    const float* p_b3   = (const float*)d_in[38];

    float* ws = (float*)d_ws;
    size_t off = 0;
    float* h      = ws + off; off += (size_t)Mrows * Dc;         // 2048x512
    float* buf1   = ws + off; off += (size_t)Mrows * DFFc;       // 2048x2048
    float* buf2   = ws + off; off += (size_t)Mrows * Vp;         // 2048x576
    float* qb     = ws + off; off += (size_t)Mrows * Dc;
    float* kb     = ws + off; off += (size_t)Mrows * Dc;
    float* vb     = ws + off; off += (size_t)Mrows * Dc;
    float* ao     = ws + off; off += (size_t)Mrows * Dc;
    float* tmpD   = ws + off; off += (size_t)Mrows * Dc;
    float* postab = ws + off; off += (size_t)Sc * Dc;            // 1024x512
    float* wcomb  = ws + off; off += (size_t)Bc * Sc * Sc;       // 2x1024x1024
    float* embpos = ws + off; off += (size_t)Mrows * Dc;
    float* padA   = ws + off; off += (size_t)Mrows * Vp;         // padded dec_in
    float* wpadA  = ws + off; off += (size_t)Vp * Vp;            // 576x576
    float* wpadB  = ws + off; off += (size_t)Vp * Vp;
    float* wpadC  = ws + off; off += (size_t)Vp * Vp;
    float* biasA  = ws + off; off += (size_t)Vp;
    float* biasB  = ws + off; off += (size_t)Vp;
    float* biasC  = ws + off; off += (size_t)Vp;
    float* smask  = ws + off; off += (size_t)Mrows;
    float* emask  = ws + off; off += (size_t)Mrows;

    // 1) sinusoid table, graph kernel combine, pad masks
    {
        int n = Sc * Dc;
        postable_kernel<<<(n + 255) / 256, 256, 0, stream>>>(postab, Sc, Dc);
    }
    {
        long long n4 = (long long)Bc * Sc * Sc / 4;
        wcomb_kernel<<<(unsigned)((n4 + 255) / 256), 256, 0, stream>>>(gk, wcomb, Bc, Sc);
    }
    rowmask_kernel<<<(Mrows + 127) / 128, 128, 0, stream>>>(dec_in, smask, Mrows, Vc);
    rowmask_kernel<<<(Mrows + 127) / 128, 128, 0, stream>>>(prev,   emask, Mrows, Dc);

    // 2) embedding MLP on padded shapes (513 -> 576, zero padding)
    launch_pad(dec_in, padA,  Mrows, Vc, Mrows, Vp, stream);
    launch_pad(emb_w1, wpadA, Vc, Vc, Vp, Vp, stream);
    launch_pad(emb_w2, wpadB, Vc, Vc, Vp, Vp, stream);
    launch_pad(emb_w3, wpadC, Vc, Dc, Vp, Dc, stream);
    launch_pad(emb_b1, biasA, 1, Vc, 1, Vp, stream);
    launch_pad(emb_b2, biasB, 1, Vc, 1, Vp, stream);
    launch_gemm(padA, wpadA, biasA, buf2, Mrows, Vp, Vp, 1, stream);
    launch_gemm(buf2, wpadB, biasB, buf1, Mrows, Vp, Vp, 1, stream);
    launch_gemm(buf1, wpadC, emb_b3, tmpD, Mrows, Dc, Vp, 0, stream);

    // 3) graph positional embedding: emb[b] = Wc[b] @ postab
    for (int b = 0; b < Bc; b++)
        launch_gemm(wcomb + (size_t)b * Sc * Sc, postab, nullptr,
                    embpos + (size_t)b * Sc * Dc, Sc, Dc, Sc, 0, stream);

    // 4) h = LN(emb + pos_emb)
    ln_kernel<<<Mrows, 128, 0, stream>>>(tmpD, embpos, ln0_g, ln0_b, h, Dc);

    // 5) decoder layers
    for (int i = 0; i < Lc; i++) {
        const size_t wOff = (size_t)i * Dc * Dc;
        // self-attention
        launch_gemm(h, wq + wOff, nullptr, qb, Mrows, Dc, Dc, 0, stream);
        launch_gemm(h, wk + wOff, nullptr, kb, Mrows, Dc, Dc, 0, stream);
        launch_gemm(h, wv + wOff, nullptr, vb, Mrows, Dc, Dc, 0, stream);
        attn_kernel<<<dim3(Sc / 64, Bc * Hc), 128, 0, stream>>>(
            qb, kb, vb, smask, ao, Bc, Sc, Hc, /*causal=*/1);
        launch_gemm(ao, wo + wOff, nullptr, tmpD, Mrows, Dc, Dc, 0, stream);
        ln_kernel<<<Mrows, 128, 0, stream>>>(h, tmpD, ln1_g + i * Dc, ln1_b + i * Dc, h, Dc);

        // cross-attention (kv from prev_dec_out)
        launch_gemm(h,    cq + wOff, nullptr, qb, Mrows, Dc, Dc, 0, stream);
        launch_gemm(prev, ck + wOff, nullptr, kb, Mrows, Dc, Dc, 0, stream);
        launch_gemm(prev, cv + wOff, nullptr, vb, Mrows, Dc, Dc, 0, stream);
        attn_kernel<<<dim3(Sc / 64, Bc * Hc), 128, 0, stream>>>(
            qb, kb, vb, emask, ao, Bc, Sc, Hc, /*causal=*/0);
        launch_gemm(ao, co + wOff, nullptr, tmpD, Mrows, Dc, Dc, 0, stream);
        ln_kernel<<<Mrows, 128, 0, stream>>>(h, tmpD, ln2_g + i * Dc, ln2_b + i * Dc, h, Dc);

        // FFN
        launch_gemm(h, ffn_w1 + (size_t)i * Dc * DFFc, ffn_b1 + (size_t)i * DFFc,
                    buf1, Mrows, DFFc, Dc, 1, stream);
        launch_gemm(buf1, ffn_w2 + (size_t)i * DFFc * Dc, ffn_b2 + (size_t)i * Dc,
                    tmpD, Mrows, Dc, DFFc, 0, stream);
        ln_kernel<<<Mrows, 128, 0, stream>>>(h, tmpD, ln3_g + i * Dc, ln3_b + i * Dc, h, Dc);
    }

    // 6) output projection MLP on padded shapes, then unpad into d_out
    launch_pad(p_w1, wpadA, Dc, Vc, Dc, Vp, stream);
    launch_pad(p_w2, wpadB, Vc, Vc, Vp, Vp, stream);
    launch_pad(p_w3, wpadC, Vc, Vc, Vp, Vp, stream);
    launch_pad(p_b1, biasA, 1, Vc, 1, Vp, stream);
    launch_pad(p_b2, biasB, 1, Vc, 1, Vp, stream);
    launch_pad(p_b3, biasC, 1, Vc, 1, Vp, stream);
    launch_gemm(h,    wpadA, biasA, buf2, Mrows, Vp, Dc, 1, stream);
    launch_gemm(buf2, wpadB, biasB, buf1, Mrows, Vp, Vp, 1, stream);
    launch_gemm(buf1, wpadC, biasC, buf2, Mrows, Vp, Vp, 0, stream);
    {
        int n = Mrows * Vc;
        unpad_copy_kernel<<<(n + 255) / 256, 256, 0, stream>>>(
            buf2, (float*)d_out, Mrows, Vp, Vc);
    }
}
